// SAModule_43911745634593
// MI455X (gfx1250) — compile-verified, matrix operand order checked by reading, not run on hardware
//
#include <hip/hip_runtime.h>
#include <math.h>

// Problem constants (match reference)
#define BB    8
#define NN    4096
#define FIN   64
#define SS    1024          // N * 0.25
#define KK    64            // max neighbors
#define HH    64
#define FOUT  128
#define RAD2  0.04f         // R*R, R=0.2
#define KPAD  96            // 67 padded up to 3 x 32 for f16 WMMA K-steps
#define CPB   8             // centroids per block in the WMMA kernel

#define W1FRAG (3 * 4 * 32 * 16)   // 6144 f16: [kstep][ntile][lane][16]
#define W2FRAG (2 * 8 * 32 * 16)   // 8192 f16
#define WFRAG  (W1FRAG + W2FRAG)   // 14336 f16 = 28672 bytes

typedef _Float16 half16 __attribute__((ext_vector_type(16)));
typedef _Float16 half8  __attribute__((ext_vector_type(8)));
typedef _Float16 half4  __attribute__((ext_vector_type(4)));
typedef float    f32x8  __attribute__((ext_vector_type(8)));
typedef float    f32x4  __attribute__((ext_vector_type(4)));

// ---------------------------------------------------------------------------
// Kernel 0: one-shot weight swizzle f32 -> f16 WMMA B-fragment layout.
//   elem e of lane L at kstep: K = kstep*32 + (e/8)*16 + (L/16)*8 + (e%8),
//   col n = ntile*16 + (L%16). Fragment = 16 contiguous f16 per lane.
// ---------------------------------------------------------------------------
__global__ __launch_bounds__(256) void wswz_kernel(const float* __restrict__ W1,
                                                   const float* __restrict__ W2,
                                                   _Float16* __restrict__ wsW) {
    const int e = blockIdx.x * 256 + threadIdx.x;
    if (e < W1FRAG) {
        const int ee = e & 15;
        const int ln = (e >> 4) & 31;
        const int nt = (e >> 9) & 3;
        const int ks = e >> 11;
        const int Kc = ks * 32 + (ee >> 3) * 16 + (ln >> 4) * 8 + (ee & 7);
        const int n  = nt * 16 + (ln & 15);
        wsW[e] = (_Float16)((Kc < FIN + 3) ? W1[Kc * HH + n] : 0.0f);
    } else if (e < WFRAG) {
        const int e2 = e - W1FRAG;
        const int ee = e2 & 15;
        const int ln = (e2 >> 4) & 31;
        const int nt = (e2 >> 9) & 7;
        const int ks = e2 >> 12;
        const int Kc = ks * 32 + (ee >> 3) * 16 + (ln >> 4) * 8 + (ee & 7);
        const int n  = nt * 16 + (ln & 15);
        wsW[e] = (_Float16)W2[Kc * FOUT + n];
    }
}

// ---------------------------------------------------------------------------
// Kernel 1: farthest point sampling. One block per cloud, 256 threads.
// Each thread owns 16 points (registers). 1023 argmax rounds.
// Tie-break: lowest global index (matches jnp.argmax).
// ---------------------------------------------------------------------------
__global__ __launch_bounds__(256) void fps_kernel(const float* __restrict__ pos,
                                                  int* __restrict__ samp) {
    const int b = blockIdx.x;
    const int t = threadIdx.x;
    const float* base = pos + (size_t)b * NN * 3;

    __shared__ float s_red[256];
    __shared__ int   s_idx[256];
    __shared__ float s_new[3];

    float px[16], py[16], pz[16], dist[16];
    const float x0 = base[0], y0 = base[1], z0 = base[2];
#pragma unroll
    for (int i = 0; i < 16; ++i) {
        const int idx = t + i * 256;
        px[i] = base[idx * 3 + 0];
        py[i] = base[idx * 3 + 1];
        pz[i] = base[idx * 3 + 2];
        const float dx = px[i] - x0, dy = py[i] - y0, dz = pz[i] - z0;
        dist[i] = dx * dx + dy * dy + dz * dz;
    }
    if (t == 0) samp[(size_t)b * SS] = 0;

    for (int s = 1; s < SS; ++s) {
        float bd = -1.0f;
        int   bi = 0;
#pragma unroll
        for (int i = 0; i < 16; ++i) {
            const int idx = t + i * 256;
            if (dist[i] > bd || (dist[i] == bd && idx < bi)) { bd = dist[i]; bi = idx; }
        }
        s_red[t] = bd;
        s_idx[t] = bi;
        __syncthreads();
        for (int off = 128; off > 0; off >>= 1) {
            if (t < off) {
                const float od = s_red[t + off];
                const int   oi = s_idx[t + off];
                if (od > s_red[t] || (od == s_red[t] && oi < s_idx[t])) {
                    s_red[t] = od; s_idx[t] = oi;
                }
            }
            __syncthreads();
        }
        if (t == 0) {
            const int mi = s_idx[0];
            samp[(size_t)b * SS + s] = mi;
            s_new[0] = base[mi * 3 + 0];
            s_new[1] = base[mi * 3 + 1];
            s_new[2] = base[mi * 3 + 2];
        }
        __syncthreads();
        const float nx = s_new[0], ny = s_new[1], nz = s_new[2];
#pragma unroll
        for (int i = 0; i < 16; ++i) {
            const float dx = px[i] - nx, dy = py[i] - ny, dz = pz[i] - nz;
            const float d = dx * dx + dy * dy + dz * dz;
            dist[i] = fminf(dist[i], d);
        }
        __syncthreads();
    }
}

// ---------------------------------------------------------------------------
// Kernel 2: radius ball query. One wave32 per centroid; ordered stream
// compaction via ballot+popcount => first K in-radius indices ascending.
// ---------------------------------------------------------------------------
__global__ __launch_bounds__(128) void ball_kernel(const float* __restrict__ pos,
                                                   const int* __restrict__ samp,
                                                   int* __restrict__ nbr,
                                                   int* __restrict__ cnts) {
    const int gw   = (blockIdx.x * 128 + threadIdx.x) >> 5;  // centroid id
    const int lane = threadIdx.x & 31;
    if (gw >= BB * SS) return;
    const int b = gw / SS;
    const float* base = pos + (size_t)b * NN * 3;
    const int ci = samp[gw];
    const float cx = base[ci * 3 + 0], cy = base[ci * 3 + 1], cz = base[ci * 3 + 2];

    int* out = nbr + (size_t)gw * KK;
    int cnt = 0;
    for (int j0 = 0; j0 < NN && cnt < KK; j0 += 32) {
        const int j = j0 + lane;
        const float dx = base[j * 3 + 0] - cx;
        const float dy = base[j * 3 + 1] - cy;
        const float dz = base[j * 3 + 2] - cz;
        const bool pred = (dx * dx + dy * dy + dz * dz) <= RAD2;
        const unsigned mask = (unsigned)__ballot(pred);
        const int pre = __popc(mask & ((1u << lane) - 1u));
        const int slot = cnt + pre;
        if (pred && slot < KK) out[slot] = j;
        cnt += __popc(mask);
    }
    cnt = min(cnt, KK);
    if (lane == 0) cnts[gw] = cnt;
    for (int k = cnt + lane; k < KK; k += 32) out[k] = 0;  // safe gather fill
}

// ---------------------------------------------------------------------------
// Kernel 3: per-centroid MLP with WMMA f32<-f16 16x16x32, masked max-agg.
// 128 threads = 4 waves; wave w owns message rows [16w,16w+16).
// GEMM1: [64 x 96(pad67)] x [96 x 64] (+b1, relu) -> LDS
// GEMM2: [64 x 64] x [64 x 128] (+b2) -> masked max over rows -> out[128].
// Weights arrive pre-swizzled (f16 fragment layout) from workspace: the
// per-block staging is a straight b128 global->LDS vector copy.
// ---------------------------------------------------------------------------
__global__ __launch_bounds__(128) void pointconv_wmma_kernel(
    const float* __restrict__ x, const float* __restrict__ pos,
    const _Float16* __restrict__ wsW,
    const float* __restrict__ b1, const float* __restrict__ b2,
    const int* __restrict__ samp, const int* __restrict__ nbr,
    const int* __restrict__ cnts, float* __restrict__ outX) {

    __shared__ __align__(64) _Float16 sW[WFRAG];                // 28 KB
    __shared__ __align__(64) _Float16 sMsg[64 * KPAD];          // 12 KB
    __shared__ __align__(64) _Float16 sHid[64 * HH];            //  8 KB
    __shared__ __align__(64) float    sB1[HH];
    __shared__ __align__(64) float    sB2[FOUT];
    __shared__ __align__(64) float    sRed[4 * FOUT];           //  2 KB

    const int tid  = threadIdx.x;
    const int w    = tid >> 5;
    const int lane = tid & 31;
    const int ml   = lane & 15;   // m (A) or n (B/D) within tile
    const int hl   = lane >> 4;   // half selector

    // ---- stage pre-swizzled weights + biases (vector copy) ----
    {
        const half8* gw = (const half8*)wsW;
        half8* lw = (half8*)sW;
#pragma unroll
        for (int i = tid; i < WFRAG / 8; i += 128) lw[i] = gw[i];
        if (tid < HH) sB1[tid] = b1[tid];
        sB2[tid] = b2[tid];
    }
    __syncthreads();

    const _Float16* sW1 = sW;
    const _Float16* sW2 = sW + W1FRAG;

    for (int c = 0; c < CPB; ++c) {
        const int cent = blockIdx.x * CPB + c;
        const int b    = cent / SS;
        const int cnt  = cnts[cent];
        const int ci   = samp[cent];
        const float* pbase = pos + (size_t)b * NN * 3;
        const float  cx = pbase[ci * 3 + 0];
        const float  cy = pbase[ci * 3 + 1];
        const float  cz = pbase[ci * 3 + 2];

        // ---- gather messages [64 x 96] into LDS as f16 (vectorized) ----
        {
            const int row = tid >> 1;             // neighbor slot 0..63
            const int h   = tid & 1;              // half of the x-features
            const int idx = nbr[(size_t)cent * KK + row];
            const f32x4* xr = (const f32x4*)(x + (size_t)(b * NN + idx) * FIN);
            _Float16* mrow = sMsg + row * KPAD;
#pragma unroll
            for (int cq = 0; cq < 8; ++cq) {
                const f32x4 v = xr[h * 8 + cq];
                half4 hv;
                hv[0] = (_Float16)v[0]; hv[1] = (_Float16)v[1];
                hv[2] = (_Float16)v[2]; hv[3] = (_Float16)v[3];
                *(half4*)(mrow + (h * 8 + cq) * 4) = hv;
            }
            if (h) {
                const float* pr = pos + (size_t)(b * NN + idx) * 3;
                half8 t0 = {};
                t0[0] = (_Float16)(pr[0] - cx);
                t0[1] = (_Float16)(pr[1] - cy);
                t0[2] = (_Float16)(pr[2] - cz);
                const half8 z = {};
                *(half8*)(mrow + 64) = t0;
                *(half8*)(mrow + 72) = z;
                *(half8*)(mrow + 80) = z;
                *(half8*)(mrow + 88) = z;
            }
        }
        __syncthreads();

        // ---- GEMM1: 16x64 per wave, K = 3 steps of 32, bias in C ----
        f32x8 acc1[4];
#pragma unroll
        for (int nt = 0; nt < 4; ++nt) {
            const float bv = sB1[nt * 16 + ml];
#pragma unroll
            for (int e = 0; e < 8; ++e) acc1[nt][e] = bv;
        }
#pragma unroll
        for (int ks = 0; ks < 3; ++ks) {
            const _Float16* rp = sMsg + (w * 16 + ml) * KPAD + ks * 32 + hl * 8;
            union { half16 v; half8 h[2]; } a;
            a.h[0] = *(const half8*)(rp);
            a.h[1] = *(const half8*)(rp + 16);
#pragma unroll
            for (int nt = 0; nt < 4; ++nt) {
                const half16 bf = *(const half16*)(sW1 + (((ks * 4 + nt) * 32) + lane) * 16);
                acc1[nt] = __builtin_amdgcn_wmma_f32_16x16x32_f16(
                    false, a.v, false, bf, (short)0, acc1[nt], false, false);
            }
        }
        // relu -> hidden LDS (f16, row-major [64 x 64])
#pragma unroll
        for (int nt = 0; nt < 4; ++nt) {
#pragma unroll
            for (int e = 0; e < 8; ++e) {
                const int row = w * 16 + e + 8 * hl;
                const int col = nt * 16 + ml;
                const float v = acc1[nt][e];
                sHid[row * HH + col] = (_Float16)(v > 0.0f ? v : 0.0f);
            }
        }
        __syncthreads();

        // ---- GEMM2: 16x128 per wave, K = 2 steps of 32, bias in C ----
        f32x8 acc2[8];
#pragma unroll
        for (int nt = 0; nt < 8; ++nt) {
            const float bv = sB2[nt * 16 + ml];
#pragma unroll
            for (int e = 0; e < 8; ++e) acc2[nt][e] = bv;
        }
#pragma unroll
        for (int ks = 0; ks < 2; ++ks) {
            const _Float16* rp = sHid + (w * 16 + ml) * HH + ks * 32 + hl * 8;
            union { half16 v; half8 h[2]; } a;
            a.h[0] = *(const half8*)(rp);
            a.h[1] = *(const half8*)(rp + 16);
#pragma unroll
            for (int nt = 0; nt < 8; ++nt) {
                const half16 bf = *(const half16*)(sW2 + (((ks * 8 + nt) * 32) + lane) * 16);
                acc2[nt] = __builtin_amdgcn_wmma_f32_16x16x32_f16(
                    false, a.v, false, bf, (short)0, acc2[nt], false, false);
            }
        }

        // ---- masked max over this wave's 16 rows, combine halves ----
#pragma unroll
        for (int nt = 0; nt < 8; ++nt) {
            float m = -INFINITY;
#pragma unroll
            for (int e = 0; e < 8; ++e) {
                const int row = w * 16 + e + 8 * hl;
                const float v = (row < cnt) ? acc2[nt][e] : -INFINITY;
                m = fmaxf(m, v);
            }
            m = fmaxf(m, __shfl_xor(m, 16, 32));
            if (hl == 0) sRed[w * FOUT + nt * 16 + ml] = m;
        }
        __syncthreads();

        // ---- cross-wave max + empty-fill, write output row ----
        {
            const int col = tid;  // 128 threads == 128 output cols
            float m = sRed[col];
            m = fmaxf(m, sRed[FOUT + col]);
            m = fmaxf(m, sRed[2 * FOUT + col]);
            m = fmaxf(m, sRed[3 * FOUT + col]);
            outX[(size_t)cent * FOUT + col] = (cnt > 0) ? m : 0.0f;
        }
        __syncthreads();
    }
}

// ---------------------------------------------------------------------------
// Kernel 4: gathered pos / batch outputs.
// ---------------------------------------------------------------------------
__global__ __launch_bounds__(256) void tail_kernel(const float* __restrict__ pos,
                                                   const int* __restrict__ samp,
                                                   float* __restrict__ outP,
                                                   int* __restrict__ outB) {
    const int i = blockIdx.x * 256 + threadIdx.x;
    if (i >= BB * SS) return;
    const int b = i / SS;
    const int g = b * NN + samp[i];
    outP[i * 3 + 0] = pos[g * 3 + 0];
    outP[i * 3 + 1] = pos[g * 3 + 1];
    outP[i * 3 + 2] = pos[g * 3 + 2];
    outB[i] = b;
}

// ---------------------------------------------------------------------------
extern "C" void kernel_launch(void* const* d_in, const int* in_sizes, int n_in,
                              void* d_out, int out_size, void* d_ws, size_t ws_size,
                              hipStream_t stream) {
    const float* x   = (const float*)d_in[0];
    const float* pos = (const float*)d_in[1];
    // d_in[2] = batch (int), unused: recomputed analytically
    const float* W1  = (const float*)d_in[3];
    const float* b1  = (const float*)d_in[4];
    const float* W2  = (const float*)d_in[5];
    const float* b2  = (const float*)d_in[6];

    // workspace: samp [B*S] | nbr [B*S*K] | cnts [B*S] | wsW [WFRAG f16]
    int* samp = (int*)d_ws;
    int* nbr  = samp + BB * SS;
    int* cnts = nbr + (size_t)BB * SS * KK;
    _Float16* wsW = (_Float16*)(cnts + BB * SS);   // 16B-aligned (2162688 B in)

    float* outX = (float*)d_out;                 // [B*S, 128]
    float* outP = outX + (size_t)BB * SS * FOUT; // [B*S, 3]
    int*   outB = (int*)(outP + (size_t)BB * SS * 3);

    wswz_kernel<<<(WFRAG + 255) / 256, 256, 0, stream>>>(W1, W2, wsW);
    fps_kernel<<<BB, 256, 0, stream>>>(pos, samp);
    ball_kernel<<<(BB * SS) / 4, 128, 0, stream>>>(pos, samp, nbr, cnts);
    pointconv_wmma_kernel<<<(BB * SS) / CPB, 128, 0, stream>>>(
        x, pos, wsW, b1, b2, samp, nbr, cnts, outX);
    tail_kernel<<<(BB * SS + 255) / 256, 256, 0, stream>>>(pos, samp, outP, outB);
}
